// QuantumQLSTM_65481071396158
// MI455X (gfx1250) — matrix operand order repaired and checked by reading, not compiled, not run on hardware
//
#include <hip/hip_runtime.h>
#include <hip/hip_bf16.h>

typedef __attribute__((ext_vector_type(16))) _Float16 v16h;
typedef __attribute__((ext_vector_type(8)))  float    v8f;

#define B_TOT 4096
#define S_LEN 512
#define D_IN  32
#define H_DIM 8
#define DH    40   // D_IN + H_DIM

// Native v_tanh_f32 on gfx1250 if the toolchain exposes it; raw v_rcp fallback.
#if __has_builtin(__builtin_amdgcn_tanhf)
__device__ __forceinline__ float ftanh(float x) { return __builtin_amdgcn_tanhf(x); }
#else
__device__ __forceinline__ float ftanh(float x) {
    float t = __expf(2.0f * x);
    return 1.0f - 2.0f * __builtin_amdgcn_rcpf(t + 1.0f);
}
#endif
// sigmoid(x) = 0.5 * tanh(x/2) + 0.5  -> 1 TRANS + 2 VALU, no division
__device__ __forceinline__ float fsig(float x) { return __builtin_fmaf(0.5f, ftanh(0.5f * x), 0.5f); }

__global__ __launch_bounds__(32) void qlstm_kernel(
    const float* __restrict__ x,
    const float* __restrict__ Wf, const float* __restrict__ bf, const float* __restrict__ pf,
    const float* __restrict__ Wi, const float* __restrict__ bi, const float* __restrict__ pi,
    const float* __restrict__ Wg, const float* __restrict__ bg, const float* __restrict__ pg,
    const float* __restrict__ Wo, const float* __restrict__ bo, const float* __restrict__ po,
    float* __restrict__ out)
{
    const int lane = threadIdx.x;       // 0..31
    const int tile = blockIdx.x;        // 0..255, 16 batch rows each
    const int row  = lane & 15;         // batch row within tile (B-operand column / acc lane)
    const int hi   = lane >> 4;         // 0: acc rows M0-7 (f,g cols) ; 1: M8-15 (i,o cols)

    // ---------------- A operands: W^T, fully loop-invariant in VGPRs ----------------
    // tile0: M0-7 = f columns, M8-15 = i columns ; tile1: g / o
    const int m    = lane & 15;                 // A-matrix row held by this lane
    const int col  = m & 7;                     // gate column
    const float* WA0 = (m < 8) ? Wf : Wi;
    const float* WA1 = (m < 8) ? Wg : Wo;
    const int base = hi ? 8 : 0;                // A-layout K base for this lane half
    v16h Aw00, Aw01, Aw10, Aw11;
#pragma unroll
    for (int idx = 0; idx < 16; ++idx) {
        int k0 = 2 * base + idx;   // K within chunk0 (contiguous: k = 16*hi + idx)
        Aw00[idx] = (_Float16)WA0[k0 * H_DIM + col];
        Aw10[idx] = (_Float16)WA1[k0 * H_DIM + col];
        int k1 = 32 + k0;                                       // chunk1: comb[32..39] = h, rest pad
        Aw01[idx] = (_Float16)((k1 < DH) ? WA0[k1 * H_DIM + col] : 0.0f);
        Aw11[idx] = (_Float16)((k1 < DH) ? WA1[k1 * H_DIM + col] : 0.0f);
    }

    // bias + phi folded per accumulator register (gate column j), per lane half
    float bs0[8], bs1[8];
#pragma unroll
    for (int j = 0; j < 8; ++j) {
        bs0[j] = hi ? (bi[j] + pi[j]) : (bf[j] + pf[j]);
        bs1[j] = hi ? (bo[j] + po[j]) : (bg[j] + pg[j]);
    }

    // ---------------- per-lane x pointer (B operand chunk0 = x^T) ----------------
    // B layout: lane<16 holds K=0..15 of column `row`, lane>=16 holds K=16..31
    const float* xrow   = x + ((size_t)(tile * 16 + row)) * S_LEN * D_IN + (hi ? 16 : 0);
    float*       outrow = out + ((size_t)(tile * 16 + row)) * S_LEN * H_DIM + hi * 4;

    // recurrent state: Bh = comb^T chunk1 (h in f16, lanes<16 idx0..7), c in f32 regs
    v16h  Bh = {};
    float cst[4] = {0.f, 0.f, 0.f, 0.f};
    float h4[4]  = {0.f, 0.f, 0.f, 0.f};

    // prefetch x_0 (16 consecutive floats per lane)
    float4 c0, c1, c2, c3;
    {
        const float4* p = (const float4*)xrow;
        c0 = p[0]; c1 = p[1]; c2 = p[2]; c3 = p[3];
    }

    for (int t = 0; t < S_LEN; ++t) {
        // ---- prefetch next step's x; clamped address -> branchless, no loop peel ----
        int tn = (t + 1 < S_LEN) ? (t + 1) : (S_LEN - 1);
        const float4* p = (const float4*)(xrow + (size_t)tn * D_IN);
        float4 n0 = p[0], n1 = p[1], n2 = p[2], n3 = p[3];

        // ---- B operand chunk0: x^T, 16 f16 per lane ----
        v16h Bx;
        Bx[0]  = (_Float16)c0.x; Bx[1]  = (_Float16)c0.y; Bx[2]  = (_Float16)c0.z; Bx[3]  = (_Float16)c0.w;
        Bx[4]  = (_Float16)c1.x; Bx[5]  = (_Float16)c1.y; Bx[6]  = (_Float16)c1.z; Bx[7]  = (_Float16)c1.w;
        Bx[8]  = (_Float16)c2.x; Bx[9]  = (_Float16)c2.y; Bx[10] = (_Float16)c2.z; Bx[11] = (_Float16)c2.w;
        Bx[12] = (_Float16)c3.x; Bx[13] = (_Float16)c3.y; Bx[14] = (_Float16)c3.z; Bx[15] = (_Float16)c3.w;

        // ---- 4x v_wmma_f32_16x16x32_f16: all 32 gate pre-activations for 16 rows ----
        v8f acc0 = {}; v8f acc1 = {};
        acc0 = __builtin_amdgcn_wmma_f32_16x16x32_f16(false, Aw00, false, Bx, (short)0, acc0, false, false);
        acc1 = __builtin_amdgcn_wmma_f32_16x16x32_f16(false, Aw10, false, Bx, (short)0, acc1, false, false);
        acc0 = __builtin_amdgcn_wmma_f32_16x16x32_f16(false, Aw01, false, Bh, (short)0, acc0, false, false);
        acc1 = __builtin_amdgcn_wmma_f32_16x16x32_f16(false, Aw11, false, Bh, (short)0, acc1, false, false);

        // ---- quantum layer: cos(z+b+phi) then IN-REGISTER prefix product (depth 3) ----
        float Q0[8], Q1[8];
#pragma unroll
        for (int j = 0; j < 8; ++j) {
            Q0[j] = __cosf(acc0[j] + bs0[j]);
            Q1[j] = __cosf(acc1[j] + bs1[j]);
        }
        Q0[1] *= Q0[0]; Q0[3] *= Q0[2]; Q0[5] *= Q0[4]; Q0[7] *= Q0[6];
        Q0[2] *= Q0[1]; Q0[3] *= Q0[1]; Q0[6] *= Q0[5]; Q0[7] *= Q0[5];
        Q0[4] *= Q0[3]; Q0[5] *= Q0[3]; Q0[6] *= Q0[3]; Q0[7] *= Q0[3];
        Q1[1] *= Q1[0]; Q1[3] *= Q1[2]; Q1[5] *= Q1[4]; Q1[7] *= Q1[6];
        Q1[2] *= Q1[1]; Q1[3] *= Q1[1]; Q1[6] *= Q1[5]; Q1[7] *= Q1[5];
        Q1[4] *= Q1[3]; Q1[5] *= Q1[3]; Q1[6] *= Q1[3]; Q1[7] *= Q1[3];

        // ---- single lane-half exchange: lanes<16 compute cols 0-3, lanes>=16 cols 4-7 ----
        float r0[4], r1[4];
#pragma unroll
        for (int j = 0; j < 4; ++j) {
            float s0 = hi ? Q0[j] : Q0[4 + j];   // send what the partner needs
            float s1 = hi ? Q1[j] : Q1[4 + j];
            r0[j] = __shfl_xor(s0, 16, 32);
            r1[j] = __shfl_xor(s1, 16, 32);
        }

        // ---- LSTM pointwise: 4 columns per lane, full-wave utilization ----
#pragma unroll
        for (int j = 0; j < 4; ++j) {
            float fq = hi ? r0[j]     : Q0[j];
            float iq = hi ? Q0[4 + j] : r0[j];
            float gq = hi ? r1[j]     : Q1[j];
            float oq = hi ? Q1[4 + j] : r1[j];
            float f = fsig(fq);
            float i = fsig(iq);
            float g = ftanh(gq);
            float o = fsig(oq);
            float c = __builtin_fmaf(f, cst[j], i * g);
            cst[j] = c;
            h4[j]  = o * ftanh(c);
        }

        // ---- coalesced output: one float4 per lane straight from registers ----
        *(float4*)(outrow + (size_t)t * H_DIM) = make_float4(h4[0], h4[1], h4[2], h4[3]);

        // ---- h feedback: regather cols 4-7 into lanes<16, repack as Bh (f16) ----
        float rh0 = __shfl_xor(h4[0], 16, 32);
        float rh1 = __shfl_xor(h4[1], 16, 32);
        float rh2 = __shfl_xor(h4[2], 16, 32);
        float rh3 = __shfl_xor(h4[3], 16, 32);
        v16h nb = {};
        nb[0] = (_Float16)(hi ? 0.f : h4[0]);
        nb[1] = (_Float16)(hi ? 0.f : h4[1]);
        nb[2] = (_Float16)(hi ? 0.f : h4[2]);
        nb[3] = (_Float16)(hi ? 0.f : h4[3]);
        nb[4] = (_Float16)(hi ? 0.f : rh0);
        nb[5] = (_Float16)(hi ? 0.f : rh1);
        nb[6] = (_Float16)(hi ? 0.f : rh2);
        nb[7] = (_Float16)(hi ? 0.f : rh3);
        Bh = nb;

        c0 = n0; c1 = n1; c2 = n2; c3 = n3;
    }

    // ---------------- final hx / cx, straight from registers ----------------
    float* hxdst = out + (size_t)B_TOT * S_LEN * H_DIM + (size_t)(tile * 16 + row) * H_DIM + hi * 4;
    float* cxdst = hxdst + (size_t)B_TOT * H_DIM;
    *(float4*)hxdst = make_float4(h4[0], h4[1], h4[2], h4[3]);
    *(float4*)cxdst = make_float4(cst[0], cst[1], cst[2], cst[3]);
}

extern "C" void kernel_launch(void* const* d_in, const int* in_sizes, int n_in,
                              void* d_out, int out_size, void* d_ws, size_t ws_size,
                              hipStream_t stream) {
    const float* x  = (const float*)d_in[0];
    const float* Wf = (const float*)d_in[1];
    const float* bf = (const float*)d_in[2];
    const float* pf = (const float*)d_in[3];
    const float* Wi = (const float*)d_in[4];
    const float* bi = (const float*)d_in[5];
    const float* pi = (const float*)d_in[6];
    const float* Wg = (const float*)d_in[7];
    const float* bg = (const float*)d_in[8];
    const float* pg = (const float*)d_in[9];
    const float* Wo = (const float*)d_in[10];
    const float* bo = (const float*)d_in[11];
    const float* po = (const float*)d_in[12];

    dim3 grid(B_TOT / 16);   // 256 independent waves, one 16-row batch tile each
    dim3 block(32);          // single wave32 per workgroup
    qlstm_kernel<<<grid, block, 0, stream>>>(x, Wf, bf, pf, Wi, bi, pi,
                                             Wg, bg, pg, Wo, bo, po,
                                             (float*)d_out);
}
